// MoE_68848325754922
// MI455X (gfx1250) — matrix operand order repaired and checked by reading, not compiled, not run on hardware
//
#include <hip/hip_runtime.h>
#include <hip/hip_bf16.h>
#include <math.h>

// Problem constants (from reference): B=4, S=2048, D=1024, DO=1024, E=20, K=2
#define D_IN   1024
#define H_G    256     // D/4 gating hidden
#define E_N    20      // experts
#define TWO_E  40
#define N_TOK  8192    // B*S
#define D_OUT  1024

typedef __attribute__((ext_vector_type(16))) __bf16 v16bf;
typedef __attribute__((ext_vector_type(8)))  __bf16 v8bf;
typedef __attribute__((ext_vector_type(4)))  __bf16 v4bf;
typedef __attribute__((ext_vector_type(2)))  __bf16 v2bf;
typedef __attribute__((ext_vector_type(8)))  float  v8f;

__device__ __forceinline__ void bf16_split(float f, __bf16 &hi, __bf16 &lo) {
  __bf16 h = (__bf16)f;          // RNE to bf16
  hi = h;
  lo = (__bf16)(f - (float)h);   // residual: ~16 extra mantissa bits
}

__device__ __forceinline__ v16bf cat8(v8bf a, v8bf b) {
  return __builtin_shufflevector(a, b, 0, 1, 2, 3, 4, 5, 6, 7,
                                        8, 9, 10, 11, 12, 13, 14, 15);
}

// ---------------------------------------------------------------------------
// Kernel 0: zero output (atomic-add target) and expert counters in workspace
// ---------------------------------------------------------------------------
__global__ void moe_zero_kernel(float* __restrict__ y, int* __restrict__ cnt, int n4) {
  int idx = blockIdx.x * blockDim.x + threadIdx.x;
  if (idx < n4) ((float4*)y)[idx] = make_float4(0.f, 0.f, 0.f, 0.f);
  if (idx < 32) cnt[idx] = 0;
}

// ---------------------------------------------------------------------------
// Kernel 1: gating MLP + softmax + top-2 + expert dispatch lists
// One block (256 threads) per token.
// ---------------------------------------------------------------------------
__global__ __launch_bounds__(256) void moe_gate_kernel(
    const float* __restrict__ x,   const float* __restrict__ wg1,
    const float* __restrict__ bg1, const float* __restrict__ wg2,
    const float* __restrict__ bg2, int* __restrict__ cnt,
    int* __restrict__ tok, float* __restrict__ gat) {
  __shared__ float xs[D_IN];
  __shared__ float hs[H_G];
  __shared__ float lg[E_N];

  const int t   = blockIdx.x;
  const int tid = threadIdx.x;

  ((float4*)xs)[tid] = ((const float4*)(x + (size_t)t * D_IN))[tid];
  __syncthreads();

  float acc = bg1[tid];
#pragma unroll 8
  for (int k = 0; k < D_IN; ++k)
    acc = fmaf(xs[k], wg1[(size_t)k * H_G + tid], acc);
  hs[tid] = 0.5f * acc * (1.0f + erff(acc * 0.70710678118654752f));
  __syncthreads();

  if (tid < E_N) {
    float a = bg2[tid];
#pragma unroll 8
    for (int k = 0; k < H_G; ++k)
      a = fmaf(hs[k], wg2[(size_t)k * TWO_E + tid], a);
    lg[tid] = a;
  }
  __syncthreads();

  if (tid == 0) {
    float m = lg[0];
#pragma unroll
    for (int e = 1; e < E_N; ++e) m = fmaxf(m, lg[e]);
    float p[E_N];
    float s = 0.f;
#pragma unroll
    for (int e = 0; e < E_N; ++e) { p[e] = expf(lg[e] - m); s += p[e]; }
    const float inv = 1.f / s;

    int i0 = 0; float v0 = p[0];
#pragma unroll
    for (int e = 1; e < E_N; ++e) if (p[e] > v0) { v0 = p[e]; i0 = e; }
    int i1 = (i0 == 0) ? 1 : 0; float v1 = p[i1];
#pragma unroll
    for (int e = 0; e < E_N; ++e) if (e != i0 && p[e] > v1) { v1 = p[e]; i1 = e; }

    int pos0 = atomicAdd(&cnt[i0], 1);
    tok[i0 * N_TOK + pos0] = t;
    gat[i0 * N_TOK + pos0] = v0 * inv;
    int pos1 = atomicAdd(&cnt[i1], 1);
    tok[i1 * N_TOK + pos1] = t;
    gat[i1 * N_TOK + pos1] = v1 * inv;
  }
}

// ---------------------------------------------------------------------------
// Kernel 2: per-expert gathered GEMM, bf16x3-split WMMA, fp32 accumulate.
// Block = 256 threads = 8 wave32, tile 64 tokens x 128 cols, K-chunks of 32.
// LDS holds pre-split bf16 (hi/lo); B stored transposed so each lane's 16
// K-values lie in two contiguous runs of 8 (pure ds_load_b128 fragment builds, zero VALU).
// Global loads for chunk k+32 are issued while chunk k computes.
// Grid: (colblocks=8, rowblocks=128, experts=20); empty blocks exit early.
// ---------------------------------------------------------------------------
__global__ __launch_bounds__(256) void moe_expert_gemm_kernel(
    const float* __restrict__ x,  const float* __restrict__ We,
    const float* __restrict__ be, const int* __restrict__ cnt,
    const int* __restrict__ tok,  const float* __restrict__ gat,
    float* __restrict__ y) {
  const int e  = blockIdx.z;
  const int rb = blockIdx.y;
  const int cb = blockIdx.x;
  const int count = cnt[e];
  if (rb * 64 >= count) return;

  // rows padded to 40 bf16 (80 B): keeps 16B alignment, staggers banks
  __shared__ __align__(16) __bf16 AHi[64][40];   // [row][k]   5 KB
  __shared__ __align__(16) __bf16 ALo[64][40];
  __shared__ __align__(16) __bf16 BtHi[128][40]; // [col][k]  10 KB (transposed)
  __shared__ __align__(16) __bf16 BtLo[128][40];
  __shared__ int   stok[64];
  __shared__ float sgat[64];

  const int tid  = threadIdx.x;
  const int lane = tid & 31;
  const int wave = tid >> 5;
  const int wm   = wave & 3;   // wave row-tile (16 rows)
  const int wn   = wave >> 2;  // wave col-group (64 cols)

  if (tid < 64) {
    int r = rb * 64 + tid;
    if (r < count) { stok[tid] = tok[e * N_TOK + r]; sgat[tid] = gat[e * N_TOK + r]; }
    else           { stok[tid] = 0;                  sgat[tid] = 0.f; }
  }
  __syncthreads();

  v8f acc0 = {}, acc1 = {}, acc2 = {}, acc3 = {};

  const float* Bbase = We + (size_t)e * D_IN * D_OUT + (size_t)cb * 128;

  // A staging ownership: row = tid>>2, K quads aq..aq+3 and aq+16..aq+19
  const int    arow = tid >> 2;
  const int    aq   = (tid & 3) * 4;
  const float* xrow = x + (size_t)stok[arow] * D_IN;

  // B staging ownership: 512 tasks (16 K-row-pairs x 32 col-quads), 2 per thread
  const int rp0 = tid >> 5,          c00 = (tid & 31) * 4;           // task tid
  const int rp1 = (tid + 256) >> 5,  c01 = (tid & 31) * 4;           // task tid+256

  // fragment addressing (wave32 WMMA bf16 layouts)
  const int mr  = lane & 15;
  const int khA = (lane >> 4) * 8;   // A: elems 0..7 -> K=khA+j, elems 8..15 -> K=khA+16+j
  const int khB = (lane >> 4) * 8;   // B: elems 0..7 -> K=khB+j, elems 8..15 -> K=khB+16+j

  // ---- prologue: global loads for chunk 0 ----
  float4 ra0 = *(const float4*)(xrow + aq);
  float4 ra1 = *(const float4*)(xrow + aq + 16);
  float4 rb00 = *(const float4*)(Bbase + (size_t)(2 * rp0)     * D_OUT + c00);
  float4 rb01 = *(const float4*)(Bbase + (size_t)(2 * rp0 + 1) * D_OUT + c00);
  float4 rb10 = *(const float4*)(Bbase + (size_t)(2 * rp1)     * D_OUT + c01);
  float4 rb11 = *(const float4*)(Bbase + (size_t)(2 * rp1 + 1) * D_OUT + c01);

  for (int kc = 0; kc < D_IN; kc += 32) {
    // ---- convert staged registers -> bf16 hi/lo, store to LDS ----
    {
      float fa0[4] = {ra0.x, ra0.y, ra0.z, ra0.w};
      float fa1[4] = {ra1.x, ra1.y, ra1.z, ra1.w};
      v4bf h0, l0, h1, l1;
#pragma unroll
      for (int j = 0; j < 4; ++j) {
        __bf16 h, l;
        bf16_split(fa0[j], h, l); h0[j] = h; l0[j] = l;
        bf16_split(fa1[j], h, l); h1[j] = h; l1[j] = l;
      }
      *(v4bf*)&AHi[arow][aq]      = h0;  *(v4bf*)&ALo[arow][aq]      = l0;
      *(v4bf*)&AHi[arow][aq + 16] = h1;  *(v4bf*)&ALo[arow][aq + 16] = l1;

      float fb00[4] = {rb00.x, rb00.y, rb00.z, rb00.w};
      float fb01[4] = {rb01.x, rb01.y, rb01.z, rb01.w};
      float fb10[4] = {rb10.x, rb10.y, rb10.z, rb10.w};
      float fb11[4] = {rb11.x, rb11.y, rb11.z, rb11.w};
#pragma unroll
      for (int j = 0; j < 4; ++j) {
        __bf16 hA, lA, hB, lB;
        bf16_split(fb00[j], hA, lA); bf16_split(fb01[j], hB, lB);
        v2bf ph = {hA, hB}, pl = {lA, lB};
        *(v2bf*)&BtHi[c00 + j][2 * rp0] = ph;
        *(v2bf*)&BtLo[c00 + j][2 * rp0] = pl;
        bf16_split(fb10[j], hA, lA); bf16_split(fb11[j], hB, lB);
        v2bf qh = {hA, hB}, ql = {lA, lB};
        *(v2bf*)&BtHi[c01 + j][2 * rp1] = qh;
        *(v2bf*)&BtLo[c01 + j][2 * rp1] = ql;
      }
    }
    __syncthreads();

    // ---- issue global loads for next chunk (hidden behind WMMA work) ----
    const int kn = kc + 32;
    if (kn < D_IN) {
      ra0  = *(const float4*)(xrow + kn + aq);
      ra1  = *(const float4*)(xrow + kn + aq + 16);
      rb00 = *(const float4*)(Bbase + (size_t)(kn + 2 * rp0)     * D_OUT + c00);
      rb01 = *(const float4*)(Bbase + (size_t)(kn + 2 * rp0 + 1) * D_OUT + c00);
      rb10 = *(const float4*)(Bbase + (size_t)(kn + 2 * rp1)     * D_OUT + c01);
      rb11 = *(const float4*)(Bbase + (size_t)(kn + 2 * rp1 + 1) * D_OUT + c01);
    }

    // ---- build fragments: pure LDS b128 loads + register concatenation ----
    const int arf = wm * 16 + mr;
    v16bf a_hi = cat8(*(const v8bf*)&AHi[arf][khA], *(const v8bf*)&AHi[arf][khA + 16]);
    v16bf a_lo = cat8(*(const v8bf*)&ALo[arf][khA], *(const v8bf*)&ALo[arf][khA + 16]);

    v16bf bh[4], bl[4];
#pragma unroll
    for (int tcol = 0; tcol < 4; ++tcol) {
      int nc = wn * 64 + tcol * 16 + mr;
      bh[tcol] = cat8(*(const v8bf*)&BtHi[nc][khB], *(const v8bf*)&BtHi[nc][khB + 16]);
      bl[tcol] = cat8(*(const v8bf*)&BtLo[nc][khB], *(const v8bf*)&BtLo[nc][khB + 16]);
    }

    // ---- 12 WMMAs, round-robin over accumulators (reuse distance 4) ----
    acc0 = __builtin_amdgcn_wmma_f32_16x16x32_bf16(false, a_hi, false, bh[0], (short)0, acc0, false, false);
    acc1 = __builtin_amdgcn_wmma_f32_16x16x32_bf16(false, a_hi, false, bh[1], (short)0, acc1, false, false);
    acc2 = __builtin_amdgcn_wmma_f32_16x16x32_bf16(false, a_hi, false, bh[2], (short)0, acc2, false, false);
    acc3 = __builtin_amdgcn_wmma_f32_16x16x32_bf16(false, a_hi, false, bh[3], (short)0, acc3, false, false);
    acc0 = __builtin_amdgcn_wmma_f32_16x16x32_bf16(false, a_hi, false, bl[0], (short)0, acc0, false, false);
    acc1 = __builtin_amdgcn_wmma_f32_16x16x32_bf16(false, a_hi, false, bl[1], (short)0, acc1, false, false);
    acc2 = __builtin_amdgcn_wmma_f32_16x16x32_bf16(false, a_hi, false, bl[2], (short)0, acc2, false, false);
    acc3 = __builtin_amdgcn_wmma_f32_16x16x32_bf16(false, a_hi, false, bl[3], (short)0, acc3, false, false);
    acc0 = __builtin_amdgcn_wmma_f32_16x16x32_bf16(false, a_lo, false, bh[0], (short)0, acc0, false, false);
    acc1 = __builtin_amdgcn_wmma_f32_16x16x32_bf16(false, a_lo, false, bh[1], (short)0, acc1, false, false);
    acc2 = __builtin_amdgcn_wmma_f32_16x16x32_bf16(false, a_lo, false, bh[2], (short)0, acc2, false, false);
    acc3 = __builtin_amdgcn_wmma_f32_16x16x32_bf16(false, a_lo, false, bh[3], (short)0, acc3, false, false);

    __syncthreads();
  }

  // ---- epilogue: y[token, col] += gate * (acc + be[e, col]) ----
  // C layout: M = i + 8*(lane>>4), N = lane&15
  const int half = lane >> 4;
#pragma unroll
  for (int tcol = 0; tcol < 4; ++tcol) {
    const v8f& a = (tcol == 0) ? acc0 : (tcol == 1) ? acc1 : (tcol == 2) ? acc2 : acc3;
    int   col  = cb * 128 + wn * 64 + tcol * 16 + mr;
    float bias = be[(size_t)e * D_OUT + col];
#pragma unroll
    for (int i = 0; i < 8; ++i) {
      int rrow = wm * 16 + half * 8 + i;
      int grow = rb * 64 + rrow;
      if (grow < count) {
        float v = sgat[rrow] * (a[i] + bias);
        atomicAdd(&y[(size_t)stok[rrow] * D_OUT + col], v);
      }
    }
  }
}

// ---------------------------------------------------------------------------
extern "C" void kernel_launch(void* const* d_in, const int* in_sizes, int n_in,
                              void* d_out, int out_size, void* d_ws, size_t ws_size,
                              hipStream_t stream) {
  const float* x   = (const float*)d_in[0];
  const float* wg1 = (const float*)d_in[1];
  const float* bg1 = (const float*)d_in[2];
  const float* wg2 = (const float*)d_in[3];
  const float* bg2 = (const float*)d_in[4];
  const float* We  = (const float*)d_in[5];
  const float* be  = (const float*)d_in[6];
  float* y = (float*)d_out;

  // workspace layout: [counters 128B][token ids E*N][gate vals E*N]
  int*   cnt = (int*)d_ws;
  int*   tok = (int*)((char*)d_ws + 128);
  float* gat = (float*)((char*)d_ws + 128 + (size_t)E_N * N_TOK * sizeof(int));

  const int n4 = (N_TOK * D_OUT) / 4;
  moe_zero_kernel<<<(n4 + 255) / 256, 256, 0, stream>>>(y, cnt, n4);
  moe_gate_kernel<<<N_TOK, 256, 0, stream>>>(x, wg1, bg1, wg2, bg2, cnt, tok, gat);
  dim3 grid(8, 128, E_N);  // colblocks x rowblocks x experts
  moe_expert_gemm_kernel<<<grid, 256, 0, stream>>>(x, We, be, cnt, tok, gat, y);
}
